// CausalRopeSelfAttention_22316650070201
// MI455X (gfx1250) — matrix operand order repaired and checked by reading, not compiled, not run on hardware
//
#include <hip/hip_runtime.h>
#include <hip/hip_bf16.h>

// ---------------------------------------------------------------------------
// Causal RoPE self-attention, MI455X (gfx1250, wave32, WMMA 16x16x32 f16).
//   B=2, S=2048, E=1024, H=16, D=64.
// Stage 1: f32 -> f16 demotion of x and weights.
// Stage 2: QKV projection (WMMA, async-LDS double-buffered) + bias + RoPE.
// Stage 3: flash attention (WMMA QK^T, online softmax, WMMA P*V).
// Stage 4: output projection (WMMA, async-LDS double-buffered) + bias -> f32.
// ---------------------------------------------------------------------------

#define EMBED  1024
#define HEADS  16
#define HDIM   64
#define BATCH  2
#define SEQ    2048
#define NROWS  (BATCH * SEQ)      // 4096

typedef _Float16 half_t;
typedef half_t v4h  __attribute__((ext_vector_type(4)));
typedef half_t v8h  __attribute__((ext_vector_type(8)));
typedef half_t v16h __attribute__((ext_vector_type(16)));
typedef float  v4f  __attribute__((ext_vector_type(4)));
typedef float  v8f  __attribute__((ext_vector_type(8)));

__device__ __forceinline__ int lane_id() { return threadIdx.x & 31; }

// LDS byte offset of a __shared__ pointer: the generic (flat) address of the
// LDS aperture carries the workgroup-relative offset in addr[31:0] (ISA 10.2).
__device__ __forceinline__ unsigned lds_off(const void* p) {
  return (unsigned)(unsigned long long)p;
}

// CDNA5 async global->LDS copy (VGLOBAL GLOBAL_LOAD_ASYNC_TO_LDS_B128,
// ASYNCcnt-tracked; no VGPR data path).  GV addressing: 64-bit vaddr, no
// saddr.  One 16B chunk per lane.
__device__ __forceinline__ void async_g2l_b128(unsigned dst_lds, const void* src) {
  asm volatile("global_load_async_to_lds_b128 %0, %1, off"
               :: "v"(dst_lds), "v"(src) : "memory");
}
__device__ __forceinline__ void wait_async() {
  asm volatile("s_wait_asynccnt 0x0" ::: "memory");
}

// A-fragment (16x32, f16).  ISA 7.12.2: lanes 0-15 hold M=lane with
// K = {0..7, 16..23}; lanes 16-31 hold K = {8..15, 24..31}.  Per lane this is
// two contiguous 16-byte chunks from an M-major LDS tile.
__device__ __forceinline__ v16h frag_a(const half_t* sm, int lda, int mbase, int k0) {
  const int l = lane_id();
  const half_t* p = sm + (mbase + (l & 15)) * lda + k0 + ((l >> 4) << 3);
  v8h lo = *(const v8h*)(p);
  v8h hi = *(const v8h*)(p + 16);
  return __builtin_shufflevector(lo, hi, 0,1,2,3,4,5,6,7,8,9,10,11,12,13,14,15);
}

// B-fragment (32x16, f16).  Lane holds column N=(lane&15) with the same K
// order as frag_a: lanes 0-15 K = {0..7, 16..23}, lanes 16-31 K = {8..15, 24..31}.
// With the B tile stored N-major in LDS this is two 16-byte chunks per lane.
__device__ __forceinline__ v16h frag_b(const half_t* sm, int ldb, int nbase, int k0) {
  const int l = lane_id();
  const half_t* p = sm + (nbase + (l & 15)) * ldb + k0 + ((l >> 4) << 3);
  v8h lo = *(const v8h*)(p);
  v8h hi = *(const v8h*)(p + 16);
  return __builtin_shufflevector(lo, hi, 0,1,2,3,4,5,6,7,8,9,10,11,12,13,14,15);
}

__device__ __forceinline__ v8f wmma_f16(v16h a, v16h b, v8f c) {
  return __builtin_amdgcn_wmma_f32_16x16x32_f16(false, a, false, b, (short)0, c,
                                                false, false);
}

// ---------------------------------------------------------------------------
// Stage 1: f32 -> f16 conversion, 4 elems/thread.
// ---------------------------------------------------------------------------
__global__ __launch_bounds__(256)
void cvt_f32_to_f16_kernel(const float* __restrict__ in, half_t* __restrict__ out,
                           int n) {
  int i = (blockIdx.x * blockDim.x + threadIdx.x) << 2;
  if (i + 3 < n) {
    v4f v = *(const v4f*)(in + i);
    v4h h;
    h[0] = (half_t)v[0]; h[1] = (half_t)v[1];
    h[2] = (half_t)v[2]; h[3] = (half_t)v[3];
    *(v4h*)(out + i) = h;
  }
}

// ---------------------------------------------------------------------------
// Stage 2: QKV projection + RoPE.
// Grid: x = row tile (64 rows of x), y = proj*16 + head (proj 0=q,1=k,2=v).
// Block: 256 threads = 8 waves; 64x64 output tile = 16 WMMA subtiles,
// 2 per wave.  K-tile = 64, double-buffered LDS filled by async DMA.
// ---------------------------------------------------------------------------
#define LDG 72   // LDS pitch (halfs) for 64-wide tiles; multiple of 8 -> 16B

__global__ __launch_bounds__(256)
void qkv_rope_kernel(const half_t* __restrict__ xh,
                     const half_t* __restrict__ wq,
                     const half_t* __restrict__ wk,
                     const half_t* __restrict__ wv,
                     const float*  __restrict__ bq,
                     const float*  __restrict__ bk,
                     const float*  __restrict__ bv,
                     half_t* __restrict__ qo,
                     half_t* __restrict__ ko,
                     half_t* __restrict__ vo) {
  __shared__ half_t As[2][64 * LDG];
  __shared__ half_t Bs[2][64 * LDG];

  const int rowbase = blockIdx.x * 64;
  const int proj    = blockIdx.y / HEADS;
  const int head    = blockIdx.y % HEADS;
  const int colbase = head * HDIM;

  const half_t* W    = (proj == 0) ? wq : (proj == 1) ? wk : wv;
  const float*  bias = (proj == 0) ? bq : (proj == 1) ? bk : bv;
  half_t*       outp = (proj == 0) ? qo : (proj == 1) ? ko : vo;

  const int t  = threadIdx.x;
  const int w  = t >> 5;
  const int l  = t & 31;
  const int sn = w & 3;           // N subtile (0..3)
  const int sm0 = (w >> 2) << 1;  // M subtiles sm0, sm0+1

  v8f acc[2];
  acc[0] = (v8f){}; acc[1] = (v8f){};

  // Each thread DMAs two 16B chunks per tile per matrix (512 chunks / tile).
  const half_t* agp[2];
  const half_t* bgp[2];
  unsigned aoff[2][2], boff[2][2];
  for (int i = 0; i < 2; ++i) {
    const int c = t + i * 256;
    const int row = c >> 3, chk = (c & 7) << 3;
    agp[i] = xh + (size_t)(rowbase + row) * EMBED + chk;
    bgp[i] = W  + (size_t)(colbase + row) * EMBED + chk;
    for (int bb = 0; bb < 2; ++bb) {
      aoff[bb][i] = lds_off(&As[bb][row * LDG + chk]);
      boff[bb][i] = lds_off(&Bs[bb][row * LDG + chk]);
    }
  }

  // Prologue: DMA tile 0.
  for (int i = 0; i < 2; ++i) {
    async_g2l_b128(aoff[0][i], agp[i]);
    async_g2l_b128(boff[0][i], bgp[i]);
  }
  wait_async();
  __syncthreads();

  for (int k0 = 0; k0 < EMBED; k0 += 64) {
    const int cur = (k0 >> 6) & 1;
    const int nxt = cur ^ 1;
    if (k0 + 64 < EMBED) {   // DMA next tile while computing this one
      for (int i = 0; i < 2; ++i) {
        async_g2l_b128(aoff[nxt][i], agp[i] + k0 + 64);
        async_g2l_b128(boff[nxt][i], bgp[i] + k0 + 64);
      }
      __builtin_prefetch(agp[0] + k0 + 128, 0, 1);   // warm L2 one tile ahead
      __builtin_prefetch(bgp[0] + k0 + 128, 0, 1);
    }
    for (int kk = 0; kk < 64; kk += 32) {
      v16h bfr = frag_b(Bs[cur], LDG, sn * 16, kk);
      v16h a0  = frag_a(As[cur], LDG, sm0 * 16, kk);
      v16h a1  = frag_a(As[cur], LDG, (sm0 + 1) * 16, kk);
      acc[0] = wmma_f16(a0, bfr, acc[0]);
      acc[1] = wmma_f16(a1, bfr, acc[1]);
    }
    wait_async();
    __syncthreads();
  }

  // Epilogue: bias, RoPE (q,k only), store f16 to [B,H,S,D].
  const int d   = sn * 16 + (l & 15);
  const float bcol = bias[colbase + d];
  // inv_freq = 10000^(-(d/2)/32) = exp(-(ln 1e4)/32 * (d>>1))
  const float inv = __expf(-0.28782313662425572f * (float)(d >> 1));
  const float sgn = (d & 1) ? 1.0f : -1.0f;

  for (int i = 0; i < 2; ++i) {
    const int sm = sm0 + i;
    for (int r = 0; r < 8; ++r) {
      const int m   = sm * 16 + ((l >> 4) << 3) + r;
      const int row = rowbase + m;
      const int bb  = row >> 11;        // / SEQ
      const int s   = row & (SEQ - 1);  // % SEQ
      float val = acc[i][r] + bcol;
      if (proj < 2) {  // RoPE for q,k
        const float ang = (float)s * inv;
        const float cs = __cosf(ang);
        const float sn_ = __sinf(ang);
        const float part = __shfl_xor(val, 1);  // even<->odd dim partner
        val = val * cs + sgn * part * sn_;
      }
      outp[((size_t)(bb * HEADS + head) * SEQ + s) * HDIM + d] = (half_t)val;
    }
  }
}

// ---------------------------------------------------------------------------
// Stage 3: flash attention.
// Grid: x = query tile (128 queries), y = b*H + h.  Block: 256 threads,
// 8 waves; wave w owns query rows [w*16, w*16+16) and ALL 64 keys/dims of a
// tile, so softmax row statistics live in registers (16-lane shfl tree).
// Q and K tiles staged by async global->LDS DMA.
// ---------------------------------------------------------------------------
#define LQK 72   // LDS pitch for 64-wide tiles

__global__ __launch_bounds__(256)
void flash_attn_kernel(const half_t* __restrict__ q,
                       const half_t* __restrict__ k,
                       const half_t* __restrict__ v,
                       half_t* __restrict__ o /* [B,S,E] f16 */) {
  __shared__ half_t Qs[128 * LQK];
  __shared__ half_t Ks[64 * LQK];
  __shared__ half_t Vs[64 * LQK];            // transposed: [d][key]
  __shared__ half_t Ps[8 * 16 * LQK];        // per-wave P staging

  const int bh = blockIdx.y;
  const int b  = bh / HEADS;
  const int h  = bh % HEADS;
  const int qbase = blockIdx.x * 128;

  const size_t hoff = (size_t)(b * HEADS + h) * SEQ * HDIM;
  const half_t* qp = q + hoff;
  const half_t* kp = k + hoff;
  const half_t* vp = v + hoff;

  const int t = threadIdx.x;
  const int w = t >> 5;
  const int l = t & 31;

  // DMA Q tile (128 x 64): 1024 16B chunks, 4 per thread.
  for (int i = 0; i < 4; ++i) {
    const int c = t + i * 256;
    const int row = c >> 3, chk = (c & 7) << 3;
    async_g2l_b128(lds_off(&Qs[row * LQK + chk]),
                   &qp[(size_t)(qbase + row) * HDIM + chk]);
  }

  float m_i[8], l_i[8];
  v8f oacc[4];
  for (int r = 0; r < 8; ++r) { m_i[r] = -1e30f; l_i[r] = 0.0f; }
  for (int dn = 0; dn < 4; ++dn) oacc[dn] = (v8f){};

  const int nkt = blockIdx.x * 2 + 2;   // causal: key tiles up to q tile end
  for (int kt = 0; kt < nkt; ++kt) {
    const int kbase = kt * 64;
    __syncthreads();   // previous iteration's reads of Ks/Vs are done
    // DMA K tile (64 x 64, [key][d] = B-tile N-major for QK^T).
    for (int i = 0; i < 2; ++i) {
      const int c = t + i * 256;
      const int row = c >> 3, chk = (c & 7) << 3;
      async_g2l_b128(lds_off(&Ks[row * LQK + chk]),
                     &kp[(size_t)(kbase + row) * HDIM + chk]);
    }
    // Stage V transposed ([d][key] = B-tile N-major for P*V).
    {
      const int key = t & 63, d0 = (t >> 6) << 4;
      v8h a0 = *(const v8h*)&vp[(size_t)(kbase + key) * HDIM + d0];
      v8h a1 = *(const v8h*)&vp[(size_t)(kbase + key) * HDIM + d0 + 8];
      for (int j = 0; j < 8; ++j) Vs[(d0 + j) * LQK + key] = a0[j];
      for (int j = 0; j < 8; ++j) Vs[(d0 + 8 + j) * LQK + key] = a1[j];
    }
    wait_async();      // K (and, first pass, Q) DMA complete
    __syncthreads();

    // Scores: 16 q rows x 64 keys per wave (4 subtiles, K=64 -> 2 wmma each).
    v8f sc[4];
    for (int sn = 0; sn < 4; ++sn) sc[sn] = (v8f){};
    v16h aq0 = frag_a(Qs, LQK, w * 16, 0);
    v16h aq1 = frag_a(Qs, LQK, w * 16, 32);
    for (int sn = 0; sn < 4; ++sn) {
      v16h b0 = frag_b(Ks, LQK, sn * 16, 0);
      v16h b1 = frag_b(Ks, LQK, sn * 16, 32);
      sc[sn] = wmma_f16(aq0, b0, sc[sn]);
      sc[sn] = wmma_f16(aq1, b1, sc[sn]);
    }

    // Online softmax (scale 1/sqrt(64); causal mask; -1e30 sentinel so a
    // fully-masked row never yields NaN corrections).
    const int qrow0 = qbase + w * 16 + ((l >> 4) << 3);
    const int kcol  = kbase + (l & 15);
    float corr[8];
    for (int r = 0; r < 8; ++r) {
      const int qg = qrow0 + r;
      float rm = -1e30f;
      for (int sn = 0; sn < 4; ++sn) {
        float val = sc[sn][r] * 0.125f;
        if (kcol + sn * 16 > qg) val = -1e30f;
        sc[sn][r] = val;
        rm = fmaxf(rm, val);
      }
      rm = fmaxf(rm, __shfl_xor(rm, 1));
      rm = fmaxf(rm, __shfl_xor(rm, 2));
      rm = fmaxf(rm, __shfl_xor(rm, 4));
      rm = fmaxf(rm, __shfl_xor(rm, 8));
      const float mnew = fmaxf(m_i[r], rm);
      corr[r] = __expf(m_i[r] - mnew);
      float rs = 0.0f;
      for (int sn = 0; sn < 4; ++sn) {
        const float p = __expf(sc[sn][r] - mnew);
        sc[sn][r] = p;
        rs += p;
      }
      rs += __shfl_xor(rs, 1);
      rs += __shfl_xor(rs, 2);
      rs += __shfl_xor(rs, 4);
      rs += __shfl_xor(rs, 8);
      l_i[r] = l_i[r] * corr[r] + rs;
      m_i[r] = mnew;
    }

    // Re-layout P (C-layout -> A-layout) through per-wave LDS.
    half_t* pw = &Ps[w * 16 * LQK];
    {
      const int mrow = (l >> 4) << 3;
      const int ncol = l & 15;
      for (int sn = 0; sn < 4; ++sn)
        for (int r = 0; r < 8; ++r)
          pw[(mrow + r) * LQK + sn * 16 + ncol] = (half_t)sc[sn][r];
    }
    asm volatile("" ::: "memory");  // same-wave DS ops are in-order in HW

    // Rescale running O, then accumulate P * V.
    for (int dn = 0; dn < 4; ++dn)
      for (int r = 0; r < 8; ++r)
        oacc[dn][r] *= corr[r];

    v16h ap0 = frag_a(pw, LQK, 0, 0);
    v16h ap1 = frag_a(pw, LQK, 0, 32);
    for (int dn = 0; dn < 4; ++dn) {
      v16h b0 = frag_b(Vs, LQK, dn * 16, 0);
      v16h b1 = frag_b(Vs, LQK, dn * 16, 32);
      oacc[dn] = wmma_f16(ap0, b0, oacc[dn]);
      oacc[dn] = wmma_f16(ap1, b1, oacc[dn]);
    }
  }

  // Epilogue: normalize, store f16 to [B,S,E] for the output projection.
  for (int dn = 0; dn < 4; ++dn) {
    const int d = dn * 16 + (l & 15);
    for (int r = 0; r < 8; ++r) {
      const int s = qbase + w * 16 + ((l >> 4) << 3) + r;
      const float ov = oacc[dn][r] / l_i[r];
      o[((size_t)(b * SEQ + s)) * EMBED + h * HDIM + d] = (half_t)ov;
    }
  }
}

// ---------------------------------------------------------------------------
// Stage 4: output projection (attn f16 @ Wo^T f16 + bo) -> f32 d_out.
// Same async double-buffered GEMM as stage 2, f32 epilogue.
// ---------------------------------------------------------------------------
__global__ __launch_bounds__(256)
void oproj_kernel(const half_t* __restrict__ ah,
                  const half_t* __restrict__ wo,
                  const float*  __restrict__ bo,
                  float* __restrict__ out) {
  __shared__ half_t As[2][64 * LDG];
  __shared__ half_t Bs[2][64 * LDG];

  const int rowbase = blockIdx.x * 64;
  const int colbase = blockIdx.y * 64;

  const int t  = threadIdx.x;
  const int w  = t >> 5;
  const int l  = t & 31;
  const int sn = w & 3;
  const int sm0 = (w >> 2) << 1;

  v8f acc[2];
  acc[0] = (v8f){}; acc[1] = (v8f){};

  const half_t* agp[2];
  const half_t* bgp[2];
  unsigned aoff[2][2], boff[2][2];
  for (int i = 0; i < 2; ++i) {
    const int c = t + i * 256;
    const int row = c >> 3, chk = (c & 7) << 3;
    agp[i] = ah + (size_t)(rowbase + row) * EMBED + chk;
    bgp[i] = wo + (size_t)(colbase + row) * EMBED + chk;
    for (int bb = 0; bb < 2; ++bb) {
      aoff[bb][i] = lds_off(&As[bb][row * LDG + chk]);
      boff[bb][i] = lds_off(&Bs[bb][row * LDG + chk]);
    }
  }

  for (int i = 0; i < 2; ++i) {
    async_g2l_b128(aoff[0][i], agp[i]);
    async_g2l_b128(boff[0][i], bgp[i]);
  }
  wait_async();
  __syncthreads();

  for (int k0 = 0; k0 < EMBED; k0 += 64) {
    const int cur = (k0 >> 6) & 1;
    const int nxt = cur ^ 1;
    if (k0 + 64 < EMBED) {
      for (int i = 0; i < 2; ++i) {
        async_g2l_b128(aoff[nxt][i], agp[i] + k0 + 64);
        async_g2l_b128(boff[nxt][i], bgp[i] + k0 + 64);
      }
      __builtin_prefetch(agp[0] + k0 + 128, 0, 1);
      __builtin_prefetch(bgp[0] + k0 + 128, 0, 1);
    }
    for (int kk = 0; kk < 64; kk += 32) {
      v16h bfr = frag_b(Bs[cur], LDG, sn * 16, kk);
      v16h a0  = frag_a(As[cur], LDG, sm0 * 16, kk);
      v16h a1  = frag_a(As[cur], LDG, (sm0 + 1) * 16, kk);
      acc[0] = wmma_f16(a0, bfr, acc[0]);
      acc[1] = wmma_f16(a1, bfr, acc[1]);
    }
    wait_async();
    __syncthreads();
  }

  const int n = sn * 16 + (l & 15);
  const float bcol = bo[colbase + n];
  for (int i = 0; i < 2; ++i) {
    for (int r = 0; r < 8; ++r) {
      const int m = (sm0 + i) * 16 + ((l >> 4) << 3) + r;
      out[(size_t)(rowbase + m) * EMBED + colbase + n] = acc[i][r] + bcol;
    }
  }
}

// ---------------------------------------------------------------------------
// Host launcher.
// ---------------------------------------------------------------------------
extern "C" void kernel_launch(void* const* d_in, const int* in_sizes, int n_in,
                              void* d_out, int out_size, void* d_ws, size_t ws_size,
                              hipStream_t stream) {
  const float* x  = (const float*)d_in[0];
  const float* Wq = (const float*)d_in[1];
  const float* bq = (const float*)d_in[2];
  const float* Wk = (const float*)d_in[3];
  const float* bk = (const float*)d_in[4];
  const float* Wv = (const float*)d_in[5];
  const float* bv = (const float*)d_in[6];
  const float* Wo = (const float*)d_in[7];
  const float* bo = (const float*)d_in[8];

  char* ws = (char*)d_ws;
  const size_t SZ_X = (size_t)NROWS * EMBED * sizeof(half_t);   // 8 MB
  const size_t SZ_W = (size_t)EMBED * EMBED * sizeof(half_t);   // 2 MB
  const size_t SZ_H = (size_t)BATCH * HEADS * SEQ * HDIM * sizeof(half_t); // 8 MB

  half_t* xh   = (half_t*)(ws);
  half_t* wqh  = (half_t*)(ws + SZ_X);
  half_t* wkh  = (half_t*)(ws + SZ_X + SZ_W);
  half_t* wvh  = (half_t*)(ws + SZ_X + 2 * SZ_W);
  half_t* woh  = (half_t*)(ws + SZ_X + 3 * SZ_W);
  half_t* qws  = (half_t*)(ws + SZ_X + 4 * SZ_W);
  half_t* kws  = (half_t*)(ws + SZ_X + 4 * SZ_W + SZ_H);
  half_t* vws  = (half_t*)(ws + SZ_X + 4 * SZ_W + 2 * SZ_H);
  half_t* aws  = (half_t*)(ws + SZ_X + 4 * SZ_W + 3 * SZ_H);

  // Stage 1: demote to f16.
  {
    const int nx = NROWS * EMBED;          // 4,194,304
    const int nw = EMBED * EMBED;          // 1,048,576
    cvt_f32_to_f16_kernel<<<nx / 1024, 256, 0, stream>>>(x, xh, nx);
    cvt_f32_to_f16_kernel<<<nw / 1024, 256, 0, stream>>>(Wq, wqh, nw);
    cvt_f32_to_f16_kernel<<<nw / 1024, 256, 0, stream>>>(Wk, wkh, nw);
    cvt_f32_to_f16_kernel<<<nw / 1024, 256, 0, stream>>>(Wv, wvh, nw);
    cvt_f32_to_f16_kernel<<<nw / 1024, 256, 0, stream>>>(Wo, woh, nw);
  }

  // Stage 2: QKV projection + RoPE.
  qkv_rope_kernel<<<dim3(NROWS / 64, 3 * HEADS), 256, 0, stream>>>(
      xh, wqh, wkh, wvh, bq, bk, bv, qws, kws, vws);

  // Stage 3: flash attention.
  flash_attn_kernel<<<dim3(SEQ / 128, BATCH * HEADS), 256, 0, stream>>>(
      qws, kws, vws, aws);

  // Stage 4: output projection.
  oproj_kernel<<<dim3(NROWS / 64, EMBED / 64), 256, 0, stream>>>(
      aws, woh, bo, (float*)d_out);
}
